// PatchTSTEncoder_7593502180054
// MI455X (gfx1250) — matrix operand order repaired and checked by reading, not compiled, not run on hardware
//
#include <hip/hip_runtime.h>
#include <cstddef>

typedef __attribute__((ext_vector_type(16))) __bf16 v16bf;
typedef __attribute__((ext_vector_type(8)))  __bf16 v8bf;
typedef __attribute__((ext_vector_type(8)))  float  v8f;
typedef __attribute__((ext_vector_type(2)))  float  v2f;
typedef __attribute__((ext_vector_type(4)))  unsigned int v4u;
typedef __attribute__((ext_vector_type(8)))  int v8i;
typedef __attribute__((ext_vector_type(4)))  int v4i;

// ---- problem constants ----
#define NTOK   28672          // 32*128*7 tokens (== 224*128 encoder rows)
#define DM     256
#define DFF    512
#define NBH    3584           // 224 * 16 heads
#define LSEQ   128

// =====================================================================
// bf16 GEMM with TDM-staged weight panel:
//   C[M×N] = A[M×K] * B^T + bias,  B stored [N×K] row-major (= torch w)
// grid = (M/64, N/64), block = 128 (4 waves). Per block: one
// TENSOR_LOAD_TO_LDS stages the 64xK B panel (row stride 2K+16 to avoid
// bank conflicts); each wave owns one 16-row M tile and computes 4 N-tiles,
// reusing its A fragment in registers and the LDS panel 4x.
// =====================================================================
__global__ __launch_bounds__(128) void k_gemm_bf16(
    const __bf16* __restrict__ A, const __bf16* __restrict__ B,
    const float* __restrict__ bias, float* __restrict__ C,
    int M, int N, int K)
{
  __shared__ __attribute__((aligned(16))) char ldsb[64 * (2 * 512 + 16)];
  const int wave = threadIdx.x >> 5;
  const int lane = threadIdx.x & 31;
  const int mr   = lane & 15;
  const int half = lane >> 4;
  const int m0 = blockIdx.x * 64 + wave * 16;
  const int n0 = blockIdx.y * 64;
  const unsigned rs = 2u * (unsigned)K + 16u;     // LDS row stride (bytes)

  if (threadIdx.x < 32) {                         // wave 0 issues the TDM
    const unsigned lds_base = (unsigned)(size_t)(void*)ldsb; // low 32b = LDS offset
    const unsigned long long ga =
        (unsigned long long)(size_t)(B + (size_t)n0 * K);
    const unsigned padi = (K == 32) ? 3u : ((K == 256) ? 6u : 7u); // 2^(p+3) B = 2K
    v4u g0; v8i g1; v4i gz = {0, 0, 0, 0};
    g0[0] = 1u;                                          // count=1, user mode
    g0[1] = lds_base;                                    // lds_addr
    g0[2] = (unsigned)(ga & 0xFFFFFFFFu);                // global_addr[31:0]
    g0[3] = (unsigned)((ga >> 32) & 0x01FFFFFFu) | (2u << 30); // addr[56:32] | type=2
    g1[0] = (int)((1u << 16) |                           // data_size = 2 bytes
                  (1u << 20) |                           // pad_enable
                  (padi << 22) |                         // pad_interval = 2K bytes
                  (3u << 25));                           // pad_amount = 4 dwords (16B)
    g1[1] = (int)(((unsigned)K & 0xFFFFu) << 16);        // tensor_dim0[15:0]
    g1[2] = (int)(((unsigned)K >> 16) | (64u << 16));    // dim0 hi | tensor_dim1=64
    g1[3] = (int)((unsigned)K << 16);                    // dim1 hi | tile_dim0=K
    g1[4] = 64;                                          // tile_dim1 = 64 rows
    g1[5] = K;                                           // tensor_dim0_stride = K
    g1[6] = 0; g1[7] = 0;
#if defined(__clang_major__) && (__clang_major__ >= 23)
    v8i gz8 = {0, 0, 0, 0, 0, 0, 0, 0};
    __builtin_amdgcn_tensor_load_to_lds(g0, g1, gz, gz, gz8, 0);
#else
    __builtin_amdgcn_tensor_load_to_lds(g0, g1, gz, gz, 0);
#endif
    __builtin_amdgcn_s_wait_tensorcnt(0);
  }
  __syncthreads();

  const __bf16* arow = A + (size_t)(m0 + mr) * K;
  v8f acc[4] = {{}, {}, {}, {}};
  for (int k0 = 0; k0 < K; k0 += 32) {
    __builtin_prefetch(arow + k0 + 128, 0, 0);           // global_prefetch_b8
    v8bf a0 = *(const v8bf*)(arow + k0 + half * 8);
    v8bf a1 = *(const v8bf*)(arow + k0 + 16 + half * 8);
    v16bf a;
    #pragma unroll
    for (int i = 0; i < 8; ++i) { a[i] = a0[i]; a[8 + i] = a1[i]; }
    const char* bbase = ldsb + (size_t)mr * rs + (size_t)(k0 + half * 8) * 2;
    #pragma unroll
    for (int nt = 0; nt < 4; ++nt) {
      const __bf16* bp = (const __bf16*)(bbase + (size_t)(nt * 16) * rs);
      v8bf b0 = *(const v8bf*)bp;
      v8bf b1 = *(const v8bf*)(bp + 16);
      v16bf b;
      #pragma unroll
      for (int i = 0; i < 8; ++i) { b[i] = b0[i]; b[8 + i] = b1[i]; }
      acc[nt] = __builtin_amdgcn_wmma_f32_16x16x32_bf16(false, a, false, b,
                                                        (short)0, acc[nt],
                                                        false, false);
    }
  }
  const int row0 = m0 + half * 8;
  #pragma unroll
  for (int nt = 0; nt < 4; ++nt) {
    const int nc = n0 + nt * 16 + mr;
    const float bv = bias ? bias[nc] : 0.0f;
    #pragma unroll
    for (int r = 0; r < 8; ++r)
      C[(size_t)(row0 + r) * N + nc] = acc[nt][r] + bv;
  }
}

// =====================================================================
// Attention scores (fp32 WMMA 16x16x4, K = dk = 16):
//   S[bh][l][m] += scale * sum_d Q[b,l,h,d] * K[b,m,h,d]
// grid = (3584, 8 ltiles), block = 256 (8 waves = 8 mtiles)
// =====================================================================
__global__ __launch_bounds__(256) void k_scores(
    const float* __restrict__ Q, const float* __restrict__ Km,
    float* __restrict__ S, float scale)
{
  const int bh = blockIdx.x, lt = blockIdx.y;
  const int mt = threadIdx.x >> 5;
  const int lane = threadIdx.x & 31;
  const int b = bh >> 4, h = bh & 15;
  const int mr = lane & 15, half = lane >> 4;
  const float* qrow = Q  + ((size_t)(b * LSEQ + lt * 16 + mr)) * DM + h * 16;
  const float* krow = Km + ((size_t)(b * LSEQ + mt * 16 + mr)) * DM + h * 16;
  v8f c = {};
  #pragma unroll
  for (int ks = 0; ks < 4; ++ks) {
    v2f a, bb;
    a[0]  = qrow[ks * 4 + half * 2];     a[1]  = qrow[ks * 4 + half * 2 + 1];
    bb[0] = krow[ks * 4 + half * 2];     bb[1] = krow[ks * 4 + half * 2 + 1];
    c = __builtin_amdgcn_wmma_f32_16x16x4_f32(false, a, false, bb,
                                              (short)0, c, false, false);
  }
  size_t base = ((size_t)bh * LSEQ + lt * 16 + half * 8) * LSEQ + mt * 16 + mr;
  #pragma unroll
  for (int r = 0; r < 8; ++r) {
    size_t i = base + (size_t)r * LSEQ;
    S[i] = S[i] + c[r] * scale;          // realformer: accumulate in place
  }
}

// ---- row softmax over 128 scores -> bf16 attn ----
__global__ __launch_bounds__(128) void k_softmax(
    const float* __restrict__ S, __bf16* __restrict__ Aout)
{
  __shared__ float red[128];
  const size_t row = blockIdx.x;
  const int t = threadIdx.x;
  float x = S[row * LSEQ + t];
  red[t] = x; __syncthreads();
  for (int s = 64; s > 0; s >>= 1) { if (t < s) red[t] = fmaxf(red[t], red[t + s]); __syncthreads(); }
  const float mx = red[0]; __syncthreads();
  const float e = __expf(x - mx);
  red[t] = e; __syncthreads();
  for (int s = 64; s > 0; s >>= 1) { if (t < s) red[t] += red[t + s]; __syncthreads(); }
  Aout[row * LSEQ + t] = (__bf16)(e / red[0]);
}

// ---- V transpose: vt[bh][d][l] = bf16(V[(b*128+l)*256 + h*16+d]) ----
__global__ void k_vt(const float* __restrict__ v, __bf16* __restrict__ vt) {
  size_t i = (size_t)blockIdx.x * blockDim.x + threadIdx.x;
  if (i < (size_t)NBH * 16 * LSEQ) {
    int l = (int)(i & 127), d = (int)((i >> 7) & 15), bh = (int)(i >> 11);
    int b = bh >> 4, h = bh & 15;
    vt[i] = (__bf16)v[((size_t)(b * LSEQ) + l) * DM + h * 16 + d];
  }
}

// =====================================================================
// O[b,l,h,:] = attn[bh](128x128, bf16) * V[b,:,h,:](128x16) — bf16 WMMA
// vt layout makes each B fragment two contiguous 16-byte loads.
// grid = 3584, block = 256 (8 waves = 8 l-tiles)
// =====================================================================
__global__ __launch_bounds__(256) void k_attnv(
    const __bf16* __restrict__ At, const __bf16* __restrict__ vt,
    float* __restrict__ O)
{
  const int bh = blockIdx.x;
  const int mt = threadIdx.x >> 5;
  const int lane = threadIdx.x & 31;
  const int b = bh >> 4, h = bh & 15;
  const int mr = lane & 15, half = lane >> 4;
  const __bf16* arow = At + ((size_t)bh * LSEQ + mt * 16 + mr) * LSEQ;
  const __bf16* brow = vt + ((size_t)bh * 16 + mr) * LSEQ;
  v8f c = {};
  for (int k0 = 0; k0 < LSEQ; k0 += 32) {
    v8bf a0 = *(const v8bf*)(arow + k0 + half * 8);
    v8bf a1 = *(const v8bf*)(arow + k0 + 16 + half * 8);
    v8bf b0 = *(const v8bf*)(brow + k0 + half * 8);
    v8bf b1 = *(const v8bf*)(brow + k0 + 16 + half * 8);
    v16bf a, bb;
    #pragma unroll
    for (int i = 0; i < 8; ++i) {
      a[i] = a0[i];  a[8 + i] = a1[i];
      bb[i] = b0[i]; bb[8 + i] = b1[i];
    }
    c = __builtin_amdgcn_wmma_f32_16x16x32_bf16(false, a, false, bb,
                                                (short)0, c, false, false);
  }
  #pragma unroll
  for (int r = 0; r < 8; ++r)
    O[((size_t)(b * LSEQ) + mt * 16 + half * 8 + r) * DM + h * 16 + mr] = c[r];
}

// ---- elementwise helpers ----
__global__ void k_cvt(const float* __restrict__ in, __bf16* __restrict__ out, size_t n) {
  size_t i = (size_t)blockIdx.x * blockDim.x + threadIdx.x;
  if (i < n) out[i] = (__bf16)in[i];
}
__global__ void k_gelu(float* __restrict__ h, __bf16* __restrict__ hb, size_t n) {
  size_t i = (size_t)blockIdx.x * blockDim.x + threadIdx.x;
  if (i < n) {
    float x = h[i];
    float g = 0.5f * x * (1.0f + erff(x * 0.70710678118654752f));  // exact gelu
    h[i] = g; hb[i] = (__bf16)g;
  }
}
__global__ void k_esq(const float* __restrict__ e, float* __restrict__ esq) {
  int k = threadIdx.x;
  float s = 0.f;
  for (int d = 0; d < DM; ++d) { float v = e[(size_t)k * DM + d]; s += v * v; }
  esq[k] = s;
}

// ---- VQ nearest-codebook + assemble encoder input u = x1 + xq + W_pos ----
__global__ __launch_bounds__(256) void k_vq(
    const float* __restrict__ x1, const float* __restrict__ x2,
    const float* __restrict__ cross, const float* __restrict__ esq,
    const float* __restrict__ vq, const float* __restrict__ wpos,
    float* __restrict__ x2_o, float* __restrict__ xq_o,
    float* __restrict__ u, __bf16* __restrict__ ub)
{
  __shared__ float sval[256]; __shared__ int sidx[256];
  const int t = blockIdx.x, k = threadIdx.x;
  sval[k] = esq[k] - 2.0f * cross[(size_t)t * 256 + k];  // ||x2||^2 constant per token
  sidx[k] = k; __syncthreads();
  for (int s = 128; s > 0; s >>= 1) {
    if (k < s) {
      float ov = sval[k + s]; int oi = sidx[k + s];
      if (ov < sval[k] || (ov == sval[k] && oi < sidx[k])) { sval[k] = ov; sidx[k] = oi; }
    }
    __syncthreads();
  }
  const int idx = sidx[0];
  const int v = t % 7, bp = t / 7, p = bp % 128, b = bp / 128;
  const float xq  = vq[(size_t)idx * DM + k];
  const float x2v = x2[(size_t)t * DM + k];
  const size_t oi = (((size_t)b * 7 + v) * 128 + p) * DM + k;   // [b][v][p][d]
  x2_o[oi] = x2v;
  xq_o[oi] = xq;
  const float uv = x1[(size_t)t * DM + k] + xq + wpos[(size_t)p * DM + k]; // STE fwd = xq
  u[oi] = uv; ub[oi] = (__bf16)uv;
}

// ---- residual add + BN batch-stat accumulation (block-reduced atomics) ----
__global__ __launch_bounds__(256) void k_resid_stats(
    const float* __restrict__ u, float* __restrict__ r, float* __restrict__ stats)
{
  const int ch = threadIdx.x;
  const size_t t0 = (size_t)blockIdx.x * 32;
  float s = 0.f, s2 = 0.f;
  for (int i = 0; i < 32; ++i) {
    size_t idx = (t0 + i) * DM + ch;
    float y = u[idx] + r[idx];
    r[idx] = y; s += y; s2 += y * y;
  }
  atomicAdd(&stats[ch], s);
  atomicAdd(&stats[256 + ch], s2);
}
__global__ __launch_bounds__(256) void k_bn_apply(
    const float* __restrict__ r, const float* __restrict__ stats,
    const float* __restrict__ g, const float* __restrict__ bb,
    float* __restrict__ u, __bf16* __restrict__ ub)
{
  const int ch = threadIdx.x;
  const size_t t0 = (size_t)blockIdx.x * 32;
  const float inv_n = 1.0f / (float)NTOK;
  const float mean = stats[ch] * inv_n;
  const float var  = stats[256 + ch] * inv_n - mean * mean;
  const float sc = rsqrtf(var + 1e-5f) * g[ch];
  const float of = bb[ch] - mean * sc;
  for (int i = 0; i < 32; ++i) {
    size_t idx = (t0 + i) * DM + ch;
    float y = r[idx] * sc + of;
    u[idx] = y; ub[idx] = (__bf16)y;
  }
}

// ---- final transpose: z[b][v][d][p] = u[(b*7+v)*128+p][d] ----
__global__ void k_final(const float* __restrict__ u, float* __restrict__ z) {
  size_t i = (size_t)blockIdx.x * blockDim.x + threadIdx.x;
  if (i < (size_t)224 * 256 * 128) {
    size_t p = i & 127, d = (i >> 7) & 255, bv = i >> 15;
    z[i] = u[(bv * 128 + p) * DM + d];
  }
}

// =====================================================================
extern "C" void kernel_launch(void* const* d_in, const int* in_sizes, int n_in,
                              void* d_out, int out_size, void* d_ws, size_t ws_size,
                              hipStream_t stream)
{
  (void)in_sizes; (void)n_in; (void)out_size; (void)ws_size;
  const float* x     = (const float*)d_in[0];
  const float* WP_w  = (const float*)d_in[1];
  const float* WP_b  = (const float*)d_in[2];
  const float* WD_w  = (const float*)d_in[3];
  const float* WD_b  = (const float*)d_in[4];
  const float* vq    = (const float*)d_in[5];
  const float* Wpos  = (const float*)d_in[6];
  const float *Wq[3], *bq[3], *Wk[3], *bk[3], *Wv[3], *bvv[3], *Wo[3], *bo[3],
              *F1[3], *fb1[3], *F2[3], *fb2[3], *g1[3], *be1[3], *g2[3], *be2[3];
  for (int i = 0; i < 3; ++i) {
    int base = 7 + i * 16;
    Wq[i]  = (const float*)d_in[base + 0];  bq[i]  = (const float*)d_in[base + 1];
    Wk[i]  = (const float*)d_in[base + 2];  bk[i]  = (const float*)d_in[base + 3];
    Wv[i]  = (const float*)d_in[base + 4];  bvv[i] = (const float*)d_in[base + 5];
    Wo[i]  = (const float*)d_in[base + 6];  bo[i]  = (const float*)d_in[base + 7];
    F1[i]  = (const float*)d_in[base + 8];  fb1[i] = (const float*)d_in[base + 9];
    F2[i]  = (const float*)d_in[base + 10]; fb2[i] = (const float*)d_in[base + 11];
    g1[i]  = (const float*)d_in[base + 12]; be1[i] = (const float*)d_in[base + 13];
    g2[i]  = (const float*)d_in[base + 14]; be2[i] = (const float*)d_in[base + 15];
  }

  float* out_z  = (float*)d_out;
  float* out_x2 = out_z  + (size_t)224 * 256 * 128;
  float* out_xq = out_x2 + (size_t)224 * 128 * 256;

  // ---- bump allocator over d_ws ----
  char* wsp = (char*)d_ws; size_t off = 0;
  auto alloc = [&](size_t bytes) -> void* {
    void* p = wsp + off; off = (off + bytes + 255) & ~(size_t)255; return p;
  };
  const size_t ND = (size_t)NTOK * DM;         // 7,340,032
  const size_t NF = (size_t)NTOK * DFF;        // 14,680,064
  const size_t NS = (size_t)NBH * LSEQ * LSEQ; // 58,720,256

  float*  S_buf  = (float*) alloc(NS * 4);
  __bf16* attn_b = (__bf16*)alloc(NS * 2);
  float*  u_f    = (float*) alloc(ND * 4);
  __bf16* u_b    = (__bf16*)alloc(ND * 2);
  float*  q_f    = (float*) alloc(ND * 4);   // aliases x1 before layers
  float*  k_f    = (float*) alloc(ND * 4);   // aliases x2 before layers
  float*  v_f    = (float*) alloc(ND * 4);
  __bf16* v_b    = (__bf16*)alloc(ND * 2);   // vt in layers; x2_bf16 before
  float*  o_f    = (float*) alloc(ND * 4);   // aliases cross before layers
  __bf16* o_b    = (__bf16*)alloc(ND * 2);
  float*  r_f    = (float*) alloc(ND * 4);
  float*  h_f    = (float*) alloc(NF * 4);
  __bf16* h_b    = (__bf16*)alloc(NF * 2);
  __bf16* x_b    = (__bf16*)alloc((size_t)NTOK * 32 * 2);
  __bf16* WPb    = (__bf16*)alloc(8192 * 2);
  __bf16* WDb    = (__bf16*)alloc(8192 * 2);
  __bf16* vqb    = (__bf16*)alloc(65536 * 2);
  __bf16 *Wqb[3], *Wkb[3], *Wvb[3], *Wob[3], *F1b[3], *F2b[3];
  for (int i = 0; i < 3; ++i) {
    Wqb[i] = (__bf16*)alloc(65536 * 2);  Wkb[i] = (__bf16*)alloc(65536 * 2);
    Wvb[i] = (__bf16*)alloc(65536 * 2);  Wob[i] = (__bf16*)alloc(65536 * 2);
    F1b[i] = (__bf16*)alloc(131072 * 2); F2b[i] = (__bf16*)alloc(131072 * 2);
  }
  float* esq   = (float*)alloc(256 * 4);
  float* stats = (float*)alloc(512 * 4);

  auto cvt = [&](const float* in, __bf16* out, size_t n) {
    k_cvt<<<dim3((unsigned)((n + 255) / 256)), 256, 0, stream>>>(in, out, n);
  };
  auto gemm = [&](const __bf16* A, const __bf16* B, const float* bias,
                  float* C, int M, int N, int K) {
    k_gemm_bf16<<<dim3(M / 64, N / 64), 128, 0, stream>>>(A, B, bias, C, M, N, K);
  };

  // ---- weight / input conversion to bf16 ----
  cvt(WP_w, WPb, 8192); cvt(WD_w, WDb, 8192); cvt(vq, vqb, 65536);
  for (int i = 0; i < 3; ++i) {
    cvt(Wq[i], Wqb[i], 65536); cvt(Wk[i], Wkb[i], 65536);
    cvt(Wv[i], Wvb[i], 65536); cvt(Wo[i], Wob[i], 65536);
    cvt(F1[i], F1b[i], 131072); cvt(F2[i], F2b[i], 131072);
  }
  cvt(x, x_b, (size_t)NTOK * 32);

  // ---- patch embeddings + VQ (token order b,p,v) ----
  float* x1_f = q_f; float* x2_f = k_f; __bf16* x2b = v_b; float* cross = o_f;
  gemm(x_b, WPb, WP_b, x1_f, NTOK, DM, 32);      // x1 = x W_P^T + b
  gemm(x_b, WDb, WD_b, x2_f, NTOK, DM, 32);      // x2 = x W_D^T + b
  cvt(x2_f, x2b, ND);
  gemm(x2b, vqb, nullptr, cross, NTOK, 256, DM); // cross = x2 . e^T
  k_esq<<<1, 256, 0, stream>>>(vq, esq);
  k_vq<<<NTOK, 256, 0, stream>>>(x1_f, x2_f, cross, esq, vq, Wpos,
                                 out_x2, out_xq, u_f, u_b);

  hipMemsetAsync(S_buf, 0, NS * 4, stream);      // realformer score accumulator

  // ---- encoder layers ----
  for (int l = 0; l < 3; ++l) {
    gemm(u_b, Wqb[l], bq[l],  q_f, NTOK, DM, DM);
    gemm(u_b, Wkb[l], bk[l],  k_f, NTOK, DM, DM);
    gemm(u_b, Wvb[l], bvv[l], v_f, NTOK, DM, DM);
    k_scores<<<dim3(NBH, 8), 256, 0, stream>>>(q_f, k_f, S_buf, 0.25f); // dk^-0.5
    k_softmax<<<(unsigned)(NBH * LSEQ), 128, 0, stream>>>(S_buf, attn_b);
    k_vt<<<(unsigned)((ND + 255) / 256), 256, 0, stream>>>(v_f, v_b);
    k_attnv<<<NBH, 256, 0, stream>>>(attn_b, v_b, o_f);
    cvt(o_f, o_b, ND);
    gemm(o_b, Wob[l], bo[l], r_f, NTOK, DM, DM);               // W_o proj
    hipMemsetAsync(stats, 0, 512 * 4, stream);
    k_resid_stats<<<NTOK / 32, 256, 0, stream>>>(u_f, r_f, stats);
    k_bn_apply<<<NTOK / 32, 256, 0, stream>>>(r_f, stats, g1[l], be1[l], u_f, u_b);
    gemm(u_b, F1b[l], fb1[l], h_f, NTOK, DFF, DM);             // FF1
    k_gelu<<<(unsigned)((NF + 255) / 256), 256, 0, stream>>>(h_f, h_b, NF);
    gemm(h_b, F2b[l], fb2[l], r_f, NTOK, DM, DFF);             // FF2
    hipMemsetAsync(stats, 0, 512 * 4, stream);
    k_resid_stats<<<NTOK / 32, 256, 0, stream>>>(u_f, r_f, stats);
    k_bn_apply<<<NTOK / 32, 256, 0, stream>>>(r_f, stats, g2[l], be2[l], u_f, u_b);
  }

  // ---- z = swapaxes(u, 2, 3) ----
  k_final<<<(unsigned)((ND + 255) / 256), 256, 0, stream>>>(u_f, out_z);
}
